// BaseNeuron_degree_feat_31026843747084
// MI455X (gfx1250) — matrix operand order repaired and checked by reading, not compiled, not run on hardware
//
#include <hip/hip_runtime.h>

#define BINS 20
#define SSZ  128
#define GAMMA 0.2f
#define BSTRIDE 132   // row stride (halves) of row-major fp16 staging tile:
                      // 264 B keeps v4h stores 8B-aligned; bank = 2k + col/2 -> conflict-free

typedef __attribute__((ext_vector_type(16))) _Float16 v16h;
typedef __attribute__((ext_vector_type(4)))  _Float16 v4h;
typedef __attribute__((ext_vector_type(8)))  float    v8f;
typedef __attribute__((ext_vector_type(4)))  float    v4f;

// ---------------------------------------------------------------------------
// Kernel 0: zero the reduction workspace (harness poisons d_ws with 0xAA).
// ---------------------------------------------------------------------------
__global__ void zero_ws_kernel(float* __restrict__ sums,
                               unsigned int* __restrict__ counts) {
    int i = blockIdx.x * blockDim.x + threadIdx.x;
    if (i < BINS * SSZ) sums[i] = 0.0f;
    if (i < BINS)       counts[i] = 0u;
}

// ---------------------------------------------------------------------------
// Kernel 1: fused spike compute + WMMA segment-sum.
//   - 256 threads = 8 wave32s; elementwise: wave w owns rows 4w..4w+3 of the
//     chunk, lane owns 4 consecutive columns -> all global traffic is b128.
//   - WMMA phase: wave w owns columns [16w,16w+16); 2x v_wmma_f32_16x16x32_f16
//     (one-hot(bins) x spikes) accumulate the 20x128 segment sum in registers.
//   - One float atomicAdd per output cell per block at the end (values are
//     exact small integers => order-independent => deterministic).
// ---------------------------------------------------------------------------
__global__ __launch_bounds__(256)
void spike_wmma_kernel(const float* __restrict__ dv,
                       const int*   __restrict__ bd,
                       const float* __restrict__ vth,
                       const float* __restrict__ tau_p,
                       float*       __restrict__ spike_out,
                       float*       __restrict__ ws_sums,
                       unsigned int* __restrict__ ws_counts,
                       int N) {
    __shared__ __align__(16) float    th_lds[BINS * SSZ];    // 10 KB threshold table
    __shared__ __align__(16) _Float16 B_lds[32 * BSTRIDE];   // 8.25 KB fp16 spike tile (row-major)
    __shared__ int          bins_s[32];
    __shared__ unsigned int cnt_lds[BINS];

    const int tid  = threadIdx.x;
    const int lane = tid & 31;
    const int wave = tid >> 5;
    const float inv_tau = 1.0f / tau_p[0];

    // preload threshold table into LDS (stays hot for whole block)
    for (int i = tid; i < BINS * SSZ; i += 256) th_lds[i] = vth[i];
    if (tid < BINS) cnt_lds[tid] = 0u;
    __syncthreads();

    // WMMA accumulators: c0 = bins 0..15, c1 = bins 16..31 (20..31 dead rows)
    v8f c0 = {}; v8f c1 = {};
    const int m0    = lane & 15;               // C layout: N = lane%16
    const int mycol = wave * 16 + m0;          // output column this lane owns
    const int k0    = (lane < 16) ? 0 : 8;     // B layout: K window per lane half
    const int col4  = lane * 4;                // elementwise: 4 cols per lane

    const int nChunks = (N + 31) >> 5;
    for (int ch = blockIdx.x; ch < nChunks; ch += gridDim.x) {
        const int row0 = ch << 5;

        // --- stage bin ids for this 32-row chunk + count rows per bin ---
        if (tid < 32) {
            int r = row0 + tid;
            int b = 0x7fffffff;                // sentinel: matches no one-hot row
            if (r < N) { b = bd[r]; atomicAdd(&cnt_lds[b], 1u); }
            bins_s[tid] = b;
        }
        __syncthreads();

        // --- elementwise (one pass): 4 rows x 4 cols per thread, b128 I/O ---
        const int4 bq = ((const int4*)bins_s)[wave];
        const int  bj[4] = {bq.x, bq.y, bq.z, bq.w};
        #pragma unroll
        for (int j = 0; j < 4; ++j) {
            const int k   = wave * 4 + j;
            const int row = row0 + k;
            v4h hv = {};                        // zeros for padded rows
            if (row < N) {                      // uniform per wave
                const v4f dvv = __builtin_nontemporal_load(
                                    (const v4f*)&dv[(size_t)row * SSZ + col4]);
                const v4f th4 = *(const v4f*)&th_lds[bj[j] * SSZ + col4];
                v4f s;
                #pragma unroll
                for (int i = 0; i < 4; ++i) {
                    s[i]  = (dvv[i] * inv_tau > th4[i]) ? 1.0f : 0.0f;
                    hv[i] = (_Float16)s[i];
                }
                __builtin_nontemporal_store(s,
                    (v4f*)&spike_out[(size_t)row * SSZ + col4]);
            }
            *(v4h*)&B_lds[k * BSTRIDE + col4] = hv;   // 8B store, conflict-free
        }
        __syncthreads();

        // --- build one-hot A tiles per ISA 16-bit A 16x32 layout ---
        // lane<16: elements 0..7 -> K 0..7,  8..15 -> K 16..23
        // lane>=16: elements 0..7 -> K 8..15, 8..15 -> K 24..31
        int kk[16];
        {
            const int4* b4 = (const int4*)bins_s;
            const int o = (lane < 16) ? 0 : 2;
            int4 q0 = b4[o], q1 = b4[o + 1], q2 = b4[o + 4], q3 = b4[o + 5];
            kk[0]=q0.x; kk[1]=q0.y; kk[2]=q0.z; kk[3]=q0.w;
            kk[4]=q1.x; kk[5]=q1.y; kk[6]=q1.z; kk[7]=q1.w;
            kk[8]=q2.x; kk[9]=q2.y; kk[10]=q2.z; kk[11]=q2.w;
            kk[12]=q3.x; kk[13]=q3.y; kk[14]=q3.z; kk[15]=q3.w;
        }
        v16h a0, a1;
        #pragma unroll
        for (int e = 0; e < 16; ++e) {
            a0[e] = (kk[e] == m0)      ? (_Float16)1.0f : (_Float16)0.0f;
            a1[e] = (kk[e] == m0 + 16) ? (_Float16)1.0f : (_Float16)0.0f;
        }

        // --- B slice (32 rows x 16 cols): lane reads its column, K window
        //     per lane half; bank = 2k + col/2 -> conflict-free d16 loads ---
        v16h b;
        #pragma unroll
        for (int e = 0; e < 16; ++e)
            b[e] = B_lds[(k0 + (e & 7) + ((e >> 3) << 4)) * BSTRIDE + mycol];

        // D = onehot(A) x spikes(B) + C  (EXEC all-ones: uniform path)
        c0 = __builtin_amdgcn_wmma_f32_16x16x32_f16(false, a0, false, b,
                                                    (short)0, c0, false, false);
        c1 = __builtin_amdgcn_wmma_f32_16x16x32_f16(false, a1, false, b,
                                                    (short)0, c1, false, false);
        __syncthreads();   // protect B_lds / bins_s before next chunk
    }

    // --- merge block partials: one atomic per (bin, col) per block ---
    #pragma unroll
    for (int j = 0; j < 8; ++j) {
        const int m = j + ((lane < 16) ? 0 : 8);   // C layout: VGPR j -> M=j / j+8
        atomicAdd(&ws_sums[m * SSZ + mycol], c0[j]);
        const int m1 = m + 16;
        if (m1 < BINS) atomicAdd(&ws_sums[m1 * SSZ + mycol], c1[j]);
    }
    if (tid < BINS) atomicAdd(&ws_counts[tid], cnt_lds[tid]);
}

// ---------------------------------------------------------------------------
// Kernel 2: v_th_new = has_nodes ? GAMMA*mean + (1-GAMMA)*th0 : th0
// ---------------------------------------------------------------------------
__global__ void finalize_kernel(const float* __restrict__ ws_sums,
                                const unsigned int* __restrict__ ws_counts,
                                const float* __restrict__ vth,
                                float* __restrict__ out_th) {
    int i = blockIdx.x * blockDim.x + threadIdx.x;
    if (i < BINS * SSZ) {
        int bin = i / SSZ;
        float cnt = (float)ws_counts[bin];
        float th0 = vth[i];
        float r = th0;
        if (cnt > 0.0f) r = GAMMA * (ws_sums[i] / cnt) + (1.0f - GAMMA) * th0;
        out_th[i] = r;
    }
}

extern "C" void kernel_launch(void* const* d_in, const int* in_sizes, int n_in,
                              void* d_out, int out_size, void* d_ws, size_t ws_size,
                              hipStream_t stream) {
    const float* dv  = (const float*)d_in[0];
    const int*   bd  = (const int*)d_in[1];
    const float* vth = (const float*)d_in[2];
    const float* tau = (const float*)d_in[3];
    // d_in[4] = alpha: unused in forward (Heaviside step)

    const int N = in_sizes[1];
    float* spike_out = (float*)d_out;
    float* out_th    = (float*)d_out + (size_t)N * SSZ;

    float*        ws_sums   = (float*)d_ws;
    unsigned int* ws_counts = (unsigned int*)((char*)d_ws + BINS * SSZ * sizeof(float));

    zero_ws_kernel<<<dim3((BINS * SSZ + 255) / 256), dim3(256), 0, stream>>>(ws_sums, ws_counts);

    const int nChunks = (N + 31) >> 5;
    const int blocks  = nChunks < 2048 ? nChunks : 2048;
    spike_wmma_kernel<<<dim3(blocks), dim3(256), 0, stream>>>(
        dv, bd, vth, tau, spike_out, ws_sums, ws_counts, N);

    finalize_kernel<<<dim3((BINS * SSZ + 255) / 256), dim3(256), 0, stream>>>(
        ws_sums, ws_counts, vth, out_th);
}
